// MultiHeadSelfAttention_74088185856342
// MI455X (gfx1250) — matrix operand order repaired and checked by reading, compiled, not run on hardware
//
#include <hip/hip_runtime.h>

typedef __attribute__((ext_vector_type(16))) __bf16 v16bf;
typedef __attribute__((ext_vector_type(8)))  __bf16 bf16x8;
typedef __attribute__((ext_vector_type(8)))  float  v8f;
typedef __attribute__((ext_vector_type(4)))  float  f32x4;
typedef __attribute__((ext_vector_type(4)))  unsigned int   u32x4;
typedef __attribute__((ext_vector_type(4)))  unsigned short u16x4;
typedef __attribute__((ext_vector_type(8)))  unsigned short u16x8;

#if __has_builtin(__builtin_amdgcn_global_load_async_to_lds_b128) && \
    __has_builtin(__builtin_amdgcn_s_wait_asynccnt)
#define HAVE_ASYNC_LDS 1   // instruction exists on this target; we emit it via asm
#endif

// ---------- helpers ----------
static __device__ __forceinline__ unsigned short f2bf(float f) {
    unsigned int u = __float_as_uint(f);
    u += 0x7FFFu + ((u >> 16) & 1u);          // round-to-nearest-even
    return (unsigned short)(u >> 16);
}
static __device__ __forceinline__ bf16x8 ld8(const unsigned short* p) {
    return *(const bf16x8*)p;                  // 16B LDS read -> ds_load_b128
}
static __device__ __forceinline__ v16bf mk16(bf16x8 lo, bf16x8 hi) {
    v16bf r;
#pragma unroll
    for (int i = 0; i < 8; ++i) { r[i] = lo[i]; r[i + 8] = hi[i]; }
    return r;
}
static __device__ __forceinline__ v8f wmma_bf16(v16bf a, v16bf b, v8f c) {
    return __builtin_amdgcn_wmma_f32_16x16x32_bf16(false, a, false, b, (short)0, c, false, false);
}
// A fragment (16x32, row-major LDS row at rowPtr): chunks at K = 8*hi and 16+8*hi
static __device__ __forceinline__ v16bf loadA(const unsigned short* rowPtr, int hi, int kbase) {
    return mk16(ld8(rowPtr + kbase + 8 * hi), ld8(rowPtr + kbase + 16 + 8 * hi));
}
// B fragment (32x16, [N][K] LDS row at rowPtr): chunks at K = 8*hi and 16+8*hi
static __device__ __forceinline__ v16bf loadB(const unsigned short* rowPtr, int hi, int kbase) {
    return mk16(ld8(rowPtr + kbase + 8 * hi), ld8(rowPtr + kbase + 16 + 8 * hi));
}
// 16B global -> LDS copy; async path (GLOBAL_LOAD_ASYNC_TO_LDS_B128, ASYNCcnt) when available.
// LDS flat addresses map by truncation, so the low 32 bits of the
// generic pointer into a __shared__ array are the wave-relative LDS byte offset for VDST.
static __device__ __forceinline__ void cp16(const unsigned short* g, unsigned short* l) {
#ifdef HAVE_ASYNC_LDS
    unsigned lds32 = (unsigned)(unsigned long long)l;
    asm volatile("global_load_async_to_lds_b128 %0, %1, off"
                 :
                 : "v"(lds32), "v"((unsigned long long)g)
                 : "memory");
#else
    *(u32x4*)l = *(const u32x4*)g;
#endif
}
static __device__ __forceinline__ void async_wait() {
#ifdef HAVE_ASYNC_LDS
    __builtin_amdgcn_s_wait_asynccnt(0);
#endif
}

// ---------- fp32 -> bf16 convert ----------
__global__ __launch_bounds__(256) void cvt_bf16(const float* __restrict__ src,
                                                unsigned short* __restrict__ dst, int n) {
    int i = (blockIdx.x * 256 + threadIdx.x) * 4;
    if (i < n) {
        f32x4 v = *(const f32x4*)&src[i];
        u16x4 o;
#pragma unroll
        for (int j = 0; j < 4; ++j) o[j] = f2bf(v[j]);
        *(u16x4*)&dst[i] = o;
    }
}

// ---------- bf16 GEMM: C(f32)[M,N] = A[M,K] * B[K,N]; 128x128 block tile ----------
// 8 waves: wave = 32(m) x 64(n) -> 8 accumulators, 8 WMMAs per 32-wide k-step.
__global__ __launch_bounds__(256) void gemm_bf16(const unsigned short* __restrict__ A,
                                                 const unsigned short* __restrict__ B,
                                                 float* __restrict__ C,
                                                 int M, int N, int K) {
    __shared__ unsigned short sA[128 * 40];   // [m][k] stride 40 (pad 8)
    __shared__ unsigned short sB[128 * 40];   // [n][k] stride 40 (pad 8)
    const int tid = threadIdx.x;
    const int m0 = blockIdx.y * 128, n0 = blockIdx.x * 128;
    const int w = tid >> 5, lane = tid & 31, ln = lane & 15, hi = lane >> 4;
    const int msub = (w & 3) * 32, nsub = (w >> 2) * 64;
    v8f acc[2][4] = {};

    for (int kk = 0; kk < K; kk += 32) {
        // stage A: 128x32 row-major, 512 x 16B chunks (async -> LDS when available)
#pragma unroll
        for (int it = 0; it < 2; ++it) {
            int id = it * 256 + tid;
            int row = id >> 2, cg = (id & 3) * 8;
            cp16(&A[(size_t)(m0 + row) * K + kk + cg], &sA[row * 40 + cg]);
        }
        // stage B transposed to [n][k]: 32x128
#pragma unroll
        for (int it = 0; it < 2; ++it) {
            int id = it * 256 + tid;
            int krow = id >> 4, ncg = (id & 15) * 8;
            u16x8 t = *(const u16x8*)&B[(size_t)(kk + krow) * N + n0 + ncg];
#pragma unroll
            for (int i = 0; i < 8; ++i) sB[(ncg + i) * 40 + krow] = t[i];
        }
        if (kk + 32 < K)
            __builtin_prefetch(&B[(size_t)(kk + 32 + (tid >> 4)) * N + n0 + (tid & 15) * 8], 0, 1);
        async_wait();
        __syncthreads();

        v16bf a0 = loadA(&sA[(msub + ln) * 40], hi, 0);
        v16bf a1 = loadA(&sA[(msub + 16 + ln) * 40], hi, 0);
#pragma unroll
        for (int nt = 0; nt < 4; ++nt) {
            v16bf b = loadB(&sB[(nsub + nt * 16 + ln) * 40], hi, 0);
            acc[0][nt] = wmma_bf16(a0, b, acc[0][nt]);
            acc[1][nt] = wmma_bf16(a1, b, acc[1][nt]);
        }
        __syncthreads();
    }
#pragma unroll
    for (int mt = 0; mt < 2; ++mt)
#pragma unroll
        for (int nt = 0; nt < 4; ++nt)
#pragma unroll
            for (int r = 0; r < 8; ++r) {
                int row = m0 + msub + mt * 16 + r + 8 * hi;
                C[(size_t)row * N + n0 + nsub + nt * 16 + ln] = acc[mt][nt][r];
            }
}

// ---------- RoPE + split into Q/K/V bf16 [B][H][T][64] ----------
__global__ __launch_bounds__(256) void rope_pack(const float* __restrict__ qkv,
                                                 unsigned short* __restrict__ Qb,
                                                 unsigned short* __restrict__ Kb,
                                                 unsigned short* __restrict__ Vb) {
    int idx = blockIdx.x * 256 + threadIdx.x;      // B*T*H*32 = 4,194,304 threads
    int i = idx & 31;
    int h = (idx >> 5) & 15;
    int t = (idx >> 9) & 2047;
    int b = idx >> 20;
    size_t row = ((size_t)b * 2048 + t) * 3072;
    int col = h * 64 + 2 * i;
    float inv = powf(10000.f, -(float)i * (1.f / 32.f));
    float ang = (float)t * inv;
    float c = cosf(ang), s = sinf(ang);
    float q0 = qkv[row + col],        q1 = qkv[row + col + 1];
    float k0 = qkv[row + 1024 + col], k1 = qkv[row + 1024 + col + 1];
    float v0 = qkv[row + 2048 + col], v1 = qkv[row + 2048 + col + 1];
    size_t ob = (((size_t)b * 16 + h) * 2048 + t) * 64 + 2 * i;
    Qb[ob]     = f2bf(q0 * c - q1 * s);
    Qb[ob + 1] = f2bf(q1 * c + q0 * s);
    Kb[ob]     = f2bf(k0 * c - k1 * s);
    Kb[ob + 1] = f2bf(k1 * c + k0 * s);
    Vb[ob]     = f2bf(v0);
    Vb[ob + 1] = f2bf(v1);
}

// ---------- flash attention: 1 block = (b,h, 128-query tile); 8 waves x 16 rows ----------
__global__ __launch_bounds__(256) void flash_attn(const unsigned short* __restrict__ Q,
                                                  const unsigned short* __restrict__ Kg,
                                                  const unsigned short* __restrict__ Vg,
                                                  const float* __restrict__ mask,
                                                  float* __restrict__ Y) {
    __shared__ unsigned short sQ[128 * 72];      // [q][d] stride 72
    __shared__ unsigned short sK[32 * 72];       // [t][d] stride 72
    __shared__ unsigned short sV[64 * 40];       // [d][t] stride 40 (transposed)
    __shared__ unsigned short sP[8 * 16 * 40];   // per-wave [16][32] stride 40
    const int T = 2048;
    const int tid = threadIdx.x;
    const int qb = blockIdx.x, bh = blockIdx.y;
    const int b = bh >> 4, h = bh & 15;
    const int q0 = qb * 128;
    const unsigned short* Qp = Q  + (size_t)bh * T * 64;
    const unsigned short* Kp = Kg + (size_t)bh * T * 64;
    const unsigned short* Vp = Vg + (size_t)bh * T * 64;

#pragma unroll
    for (int it = 0; it < 4; ++it) {             // stage 128x64 Q tile
        int id = it * 256 + tid;
        int row = id >> 3, cg = (id & 7) * 8;
        cp16(&Qp[(size_t)(q0 + row) * 64 + cg], &sQ[row * 72 + cg]);
    }
    const int w = tid >> 5, lane = tid & 31, ln = lane & 15, hi = lane >> 4;
    const int qr0 = w * 16;
    const int vrow = tid >> 3, vcg = (tid & 7) * 8;
    unsigned short* myP = &sP[w * 16 * 40];
    float m[8], l[8];
    v8f o0 = {}, o1 = {}, o2 = {}, o3 = {};
#pragma unroll
    for (int r = 0; r < 8; ++r) { m[r] = -1e30f; l[r] = 0.f; }

    const int nkv = qb * 4 + 4;                  // causal bound: kv blocks of 32
    for (int kv = 0; kv < nkv; ++kv) {
        const int kt0 = kv * 32;
        // stage K [t][d] (async path when available)
        cp16(&Kp[(size_t)(kt0 + vrow) * 64 + vcg], &sK[vrow * 72 + vcg]);
        // stage V transposed [d][t]
        {
            u16x8 t = *(const u16x8*)&Vp[(size_t)(kt0 + vrow) * 64 + vcg];
#pragma unroll
            for (int i = 0; i < 8; ++i) sV[(vcg + i) * 40 + vrow] = t[i];
        }
        async_wait();
        __syncthreads();

        // S = Q * K^T  (16 rows x 32 cols per wave)
        v8f s0 = {}, s1 = {};
        const unsigned short* qr  = &sQ[(qr0 + ln) * 72];
        const unsigned short* kr0 = &sK[ln * 72];
        const unsigned short* kr1 = &sK[(16 + ln) * 72];
#pragma unroll
        for (int ks = 0; ks < 2; ++ks) {
            v16bf a   = loadA(qr, hi, ks * 32);
            v16bf bb0 = loadB(kr0, hi, ks * 32);
            v16bf bb1 = loadB(kr1, hi, ks * 32);
            s0 = wmma_bf16(a, bb0, s0);
            s1 = wmma_bf16(a, bb1, s1);
        }

        // online softmax (rows confined to 16-lane halves)
        const int kg0 = kt0 + ln, kg1 = kt0 + 16 + ln;
        const float msk0 = (mask[b * T + kg0] - 1.f) * 1e10f;
        const float msk1 = (mask[b * T + kg1] - 1.f) * 1e10f;
#pragma unroll
        for (int r = 0; r < 8; ++r) {
            int qg = q0 + qr0 + r + 8 * hi;
            float v0 = s0[r] * 0.125f + msk0; if (kg0 > qg) v0 = -1e30f;
            float v1 = s1[r] * 0.125f + msk1; if (kg1 > qg) v1 = -1e30f;
            float mm = fmaxf(v0, v1);
#pragma unroll
            for (int off = 8; off > 0; off >>= 1) mm = fmaxf(mm, __shfl_xor(mm, off));
            float mn = fmaxf(m[r], mm);
            float al = __expf(m[r] - mn);
            float p0 = __expf(v0 - mn), p1 = __expf(v1 - mn);
            float rs = p0 + p1;
#pragma unroll
            for (int off = 8; off > 0; off >>= 1) rs += __shfl_xor(rs, off);
            m[r] = mn;
            l[r] = l[r] * al + rs;
            o0[r] *= al; o1[r] *= al; o2[r] *= al; o3[r] *= al;
            int prow = r + 8 * hi;
            myP[prow * 40 + ln]      = f2bf(p0);
            myP[prow * 40 + 16 + ln] = f2bf(p1);
        }

        // O += P * V   (P 16x32 as A-frag from per-wave LDS; V [d][t] as B-frag)
        v16bf pa = loadA(&myP[ln * 40], hi, 0);
        o0 = wmma_bf16(pa, loadB(&sV[(0  + ln) * 40], hi, 0), o0);
        o1 = wmma_bf16(pa, loadB(&sV[(16 + ln) * 40], hi, 0), o1);
        o2 = wmma_bf16(pa, loadB(&sV[(32 + ln) * 40], hi, 0), o2);
        o3 = wmma_bf16(pa, loadB(&sV[(48 + ln) * 40], hi, 0), o3);
        __syncthreads();
    }

    // Y[b][t][h*64+d] = O / l
#pragma unroll
    for (int r = 0; r < 8; ++r) {
        int qg = q0 + qr0 + r + 8 * hi;
        size_t base = ((size_t)b * T + qg) * 1024 + h * 64;
        float il = 1.f / l[r];
        Y[base + 0  + ln] = o0[r] * il;
        Y[base + 16 + ln] = o1[r] * il;
        Y[base + 32 + ln] = o2[r] * il;
        Y[base + 48 + ln] = o3[r] * il;
    }
}

extern "C" void kernel_launch(void* const* d_in, const int* in_sizes, int n_in,
                              void* d_out, int out_size, void* d_ws, size_t ws_size,
                              hipStream_t stream) {
    const float* x         = (const float*)d_in[0];   // [4,2048,1024]
    const float* attn_mask = (const float*)d_in[1];   // [4,2048]
    const float* w_qkv     = (const float*)d_in[2];   // [1024,3072]
    const float* w_proj    = (const float*)d_in[3];   // [1024,1024]
    (void)in_sizes; (void)n_in; (void)out_size; (void)ws_size;

    char* ws = (char*)d_ws;
    unsigned short* xb  = (unsigned short*)(ws);                 // 16 MB
    unsigned short* wqb = (unsigned short*)(ws + 16777216);      //  6 MB
    unsigned short* wpb = (unsigned short*)(ws + 23068672);      //  2 MB
    unsigned short* Qb  = (unsigned short*)(ws + 25165824);      // 16 MB
    unsigned short* Kb  = (unsigned short*)(ws + 41943040);      // 16 MB
    unsigned short* Vb  = (unsigned short*)(ws + 58720256);      // 16 MB
    float*          qkv = (float*)(ws + 75497472);               // 96 MB
    float*          Yf  = (float*)(ws + 75497472);               // aliases qkv (dead by then)
    unsigned short* Yb  = (unsigned short*)(ws + 75497472 + 33554432);
    float*          out = (float*)d_out;

    cvt_bf16<<<8388608 / 1024, 256, 0, stream>>>(x, xb, 8388608);
    cvt_bf16<<<3145728 / 1024, 256, 0, stream>>>(w_qkv, wqb, 3145728);
    cvt_bf16<<<1048576 / 1024, 256, 0, stream>>>(w_proj, wpb, 1048576);

    gemm_bf16<<<dim3(3072 / 128, 8192 / 128), 256, 0, stream>>>(xb, wqb, qkv, 8192, 3072, 1024);
    rope_pack<<<4194304 / 256, 256, 0, stream>>>(qkv, Qb, Kb, Vb);
    flash_attn<<<dim3(2048 / 128, 64), 256, 0, stream>>>(Qb, Kb, Vb, attn_mask, Yf);

    cvt_bf16<<<8388608 / 1024, 256, 0, stream>>>(Yf, Yb, 8388608);
    gemm_bf16<<<dim3(1024 / 128, 8192 / 128), 256, 0, stream>>>(Yb, wpb, out, 8192, 1024, 1024);
}
